// Memory_70042326663191
// MI455X (gfx1250) — matrix operand order repaired and checked by reading, not compiled, not run on hardware
//
#include <hip/hip_runtime.h>
#include <stdint.h>

// ---------------- problem constants ----------------
#define BATCH   64
#define DIMS    512
#define HW      1024            // 32*32
#define NSAMP   65536           // BATCH*HW
#define MSLOT   512

// ---------------- output offsets (floats) ----------------
#define OFF_UQ  0u                       // updated_query  [64,1024,32,32] = 67108864
#define OFF_UM  67108864u                // updated_memory [512,512]
#define OFF_SQ  67371008u                // score_query    [65536,512]
#define OFF_SM  100925440u               // score_memory   [65536,512]
#define OFF_GL  134479872u               // gathering loss
#define OFF_SL  134479873u               // spreading loss

// ---------------- workspace offsets (bytes) ----------------
#define WS_QF    0ull                    // float  [N,512]   128 MB
#define WS_QBF   134217728ull            // bf16   [N,512]    64 MB
#define WS_SCORE 201326592ull            // float  [N,512]   128 MB
#define WS_SMBF  335544320ull            // bf16   [N,512]    64 MB
#define WS_KBF   402653184ull            // bf16   [512,512] 512 KB (keys row-major)
#define WS_KTBF  403177472ull            // bf16   [512,512] 512 KB (keys^T row-major)
#define WS_CMAX  403701760ull            // uint   [512] transformed col max
#define WS_CSUM  403703808ull            // float  [512] col sum
#define WS_GIDX  403705856ull            // int    [N]
#define WS_RMAX  403968000ull            // float  [N]
#define WS_QUPD  404230144ull            // float  [512,512] 1 MB

typedef __attribute__((ext_vector_type(16))) __bf16 v16bf;
typedef __attribute__((ext_vector_type(8)))  float  v8f;

union FragBF { unsigned u[8]; v16bf v; };

static __device__ __forceinline__ unsigned short f2bf(float f) {
    unsigned u = __float_as_uint(f);
    unsigned r = u + 0x7FFFu + ((u >> 16) & 1u);   // round-to-nearest-even
    return (unsigned short)(r >> 16);
}
static __device__ __forceinline__ unsigned fflip(float f) {
    unsigned u = __float_as_uint(f);
    return (u & 0x80000000u) ? ~u : (u | 0x80000000u);
}
static __device__ __forceinline__ float funflip(unsigned u) {
    unsigned v = (u & 0x80000000u) ? (u & 0x7FFFFFFFu) : ~u;
    return __uint_as_float(v);
}

// ---------------- K0: keys -> bf16 row-major + bf16 transposed ----------------
__global__ __launch_bounds__(256)
void k_prep_keys(const float* __restrict__ keys,
                 unsigned short* __restrict__ kb,
                 unsigned short* __restrict__ ktb) {
    int i = blockIdx.x * 256 + threadIdx.x;          // 262144 total
    int r = i >> 9, c = i & 511;
    unsigned short h = f2bf(keys[i]);
    kb[i] = h;
    ktb[c * 512 + r] = h;
}

// ---------------- K1: channel L2-normalize, emit qf32 / qbf16 / out first half --
__global__ __launch_bounds__(256)
void k_normalize(const float* __restrict__ q,
                 float* __restrict__ qf,
                 unsigned short* __restrict__ qbf,
                 float* __restrict__ outUQ) {
    int b = blockIdx.x >> 2;
    int p = ((blockIdx.x & 3) << 8) + threadIdx.x;   // spatial index 0..1023
    size_t qbase = (size_t)b * (DIMS * HW) + p;
    float s = 0.f;
    for (int c = 0; c < DIMS; ++c) {
        float v = q[qbase + (size_t)c * HW];
        s += v * v;
    }
    float inv = 1.f / fmaxf(sqrtf(s), 1e-12f);
    size_t n = (size_t)b * HW + p;
    for (int c = 0; c < DIMS; ++c) {
        float v = q[qbase + (size_t)c * HW] * inv;
        qf[n * DIMS + c]  = v;
        qbf[n * DIMS + c] = f2bf(v);
        outUQ[(size_t)b * 1048576 + (size_t)c * 1024 + p] = v;  // [b, c, y, x]
    }
}

// ---------------- WMMA GEMM: C[N,512] = A[N,512]_bf16 * B^T (B rows contiguous) --
// mode 0: write score row-major AND fold per-column max into cmaxu (for the
//         axis-0 softmax) via LDS reduction + one global atomic per column.
// mode 1: scatter into updated_query[:, 512:, :, :]
__global__ __launch_bounds__(256)
void k_gemm_bf16(const unsigned short* __restrict__ A,
                 const unsigned short* __restrict__ Brows,
                 float* __restrict__ outPlain,
                 float* __restrict__ outUQ,
                 unsigned* __restrict__ cmaxu,
                 int mode) {
    __shared__ unsigned smax[64];
    const int wave = threadIdx.x >> 5;
    const int lane = threadIdx.x & 31;
    const int g    = lane >> 4;          // lane group
    const int mr   = lane & 15;          // row-in-tile / col-in-tile
    const int r0   = blockIdx.x * 128 + wave * 16;   // 16 rows per wave
    const int c0   = blockIdx.y * 64;                // 64 cols per block

    if (mode == 0) {
        if (threadIdx.x < 64) smax[threadIdx.x] = 0u;   // transformed -inf
        __syncthreads();
    }

    v8f acc[4];
    for (int t = 0; t < 4; ++t)
        for (int j = 0; j < 8; ++j) acc[t][j] = 0.f;

    for (int kb = 0; kb < 16; ++kb) {               // K = 512, 32 per step
        // A fragment: 2x contiguous 16B loads (ISA 16-bit A 16x32 lane layout)
        FragBF a;
        size_t baseA = (size_t)(r0 + mr) * DIMS + kb * 32 + g * 8;
        {
            const uint4 lo = *reinterpret_cast<const uint4*>(A + baseA);
            const uint4 hi = *reinterpret_cast<const uint4*>(A + baseA + 16);
            a.u[0] = lo.x; a.u[1] = lo.y; a.u[2] = lo.z; a.u[3] = lo.w;
            a.u[4] = hi.x; a.u[5] = hi.y; a.u[6] = hi.z; a.u[7] = hi.w;
        }
        // pull next k-slab of A toward the WGP (global_prefetch_b8)
        if (kb < 15)
            __builtin_prefetch((const void*)(A + baseA + 32), 0, 3);
        for (int t = 0; t < 4; ++t) {
            // B fragment: lane's column row in Brows, 2x 16B (same k order as A)
            FragBF bfr;
            size_t baseB = (size_t)(c0 + t * 16 + mr) * DIMS + kb * 32 + g * 8;
            const uint4 lo = *reinterpret_cast<const uint4*>(Brows + baseB);
            const uint4 hi = *reinterpret_cast<const uint4*>(Brows + baseB + 16);
            bfr.u[0] = lo.x; bfr.u[1] = lo.y; bfr.u[2] = lo.z; bfr.u[3] = lo.w;
            bfr.u[4] = hi.x; bfr.u[5] = hi.y; bfr.u[6] = hi.z; bfr.u[7] = hi.w;
            acc[t] = __builtin_amdgcn_wmma_f32_16x16x32_bf16(
                false, a.v, false, bfr.v, (short)0, acc[t], false, false);
        }
    }

    // epilogue: D lane layout: VGPR j -> row (j + 8*g), col (lane&15)
    for (int t = 0; t < 4; ++t) {
        int col = c0 + t * 16 + mr;
        float cmax = -3.4e38f;
        for (int j = 0; j < 8; ++j) {
            int row = r0 + j + 8 * g;
            float v = acc[t][j];
            cmax = fmaxf(cmax, v);
            if (mode == 0) {
                outPlain[(size_t)row * MSLOT + col] = v;
            } else {
                int b = row >> 10, p = row & 1023;
                outUQ[(size_t)b * 1048576 + (size_t)(512 + col) * 1024 + p] = v;
            }
        }
        if (mode == 0) {
            // lanes l and l+16 hold the same column (rows 0-7 vs 8-15)
            cmax = fmaxf(cmax, __shfl_xor(cmax, 16));
            if (lane < 16) atomicMax(&smax[t * 16 + mr], fflip(cmax));
        }
    }
    if (mode == 0) {
        __syncthreads();
        if (threadIdx.x < 64)
            atomicMax(&cmaxu[c0 + threadIdx.x], smax[threadIdx.x]);
    }
}

// ---------------- K3: per-row pass — softmax(axis=1), top2, losses ------------
__global__ __launch_bounds__(256)
void k_rowpass(const float* __restrict__ score,
               const float* __restrict__ qf,
               const float* __restrict__ keys,
               float* __restrict__ outSM,
               unsigned short* __restrict__ smbf,
               int*   __restrict__ gidx,
               float* __restrict__ rmax,
               float* __restrict__ gl,
               float* __restrict__ sl) {
    const int wave = threadIdx.x >> 5;
    const int lane = threadIdx.x & 31;
    const int n = blockIdx.x * 8 + wave;

    float s[16];
    for (int i = 0; i < 16; ++i)
        s[i] = score[(size_t)n * MSLOT + lane + 32 * i];

    // lane-local top2
    float v1 = -3.4e38f, v2 = -3.4e38f; int i1 = 0, i2 = 0;
    for (int i = 0; i < 16; ++i) {
        int c = lane + 32 * i; float v = s[i];
        if (v > v1 || (v == v1 && c < i1)) { v2 = v1; i2 = i1; v1 = v; i1 = c; }
        else if (v > v2 || (v == v2 && c < i2)) { v2 = v; i2 = c; }
    }
    // butterfly merge across 32 lanes
    for (int off = 16; off > 0; off >>= 1) {
        float ov1 = __shfl_xor(v1, off); int oi1 = __shfl_xor(i1, off);
        float ov2 = __shfl_xor(v2, off); int oi2 = __shfl_xor(i2, off);
        if (ov1 > v1 || (ov1 == v1 && oi1 < i1)) { v2 = v1; i2 = i1; v1 = ov1; i1 = oi1; }
        else if (ov1 > v2 || (ov1 == v2 && oi1 < i2)) { v2 = ov1; i2 = oi1; }
        if (ov2 > v2 || (ov2 == v2 && oi2 < i2)) { v2 = ov2; i2 = oi2; }
    }

    float se = 0.f;
    for (int i = 0; i < 16; ++i) se += expf(s[i] - v1);
    for (int off = 16; off > 0; off >>= 1) se += __shfl_xor(se, off);

    float gsum = 0.f, dp = 0.f, dn = 0.f;
    for (int i = 0; i < 16; ++i) {
        int c = lane + 32 * i;
        float p = expf(s[i] - v1) / se;
        outSM[(size_t)n * MSLOT + c] = p;
        smbf[(size_t)n * MSLOT + c]  = f2bf(p);
        float qv = qf[(size_t)n * DIMS + c];
        float kg = keys[(size_t)i1 * DIMS + c];
        float kn = keys[(size_t)i2 * DIMS + c];
        float dg = qv - kg;          gsum += dg * dg;
        float ep = qv - kg + 1e-6f;  dp   += ep * ep;
        float en = qv - kn + 1e-6f;  dn   += en * en;
    }
    for (int off = 16; off > 0; off >>= 1) {
        gsum += __shfl_xor(gsum, off);
        dp   += __shfl_xor(dp, off);
        dn   += __shfl_xor(dn, off);
    }
    if (lane == 0) {
        atomicAdd(gl, gsum * (1.0f / ((float)NSAMP * (float)DIMS)));
        float sp = fmaxf(sqrtf(dp) - sqrtf(dn) + 1.0f, 0.0f);
        atomicAdd(sl, sp * (1.0f / (float)NSAMP));
        gidx[n] = i1;
        rmax[n] = v1;
    }
}

// ---------------- K4b: column sum of exp(score - colmax) ----------------------
__global__ __launch_bounds__(512)
void k_colsum(const float* __restrict__ score, const unsigned* __restrict__ cmaxu,
              float* __restrict__ csum) {
    int c = threadIdx.x;
    int r0 = blockIdx.x * 256;
    float cm = funflip(cmaxu[c]);
    float s = 0.f;
    for (int r = 0; r < 256; ++r)
        s += expf(score[(size_t)(r0 + r) * MSLOT + c] - cm);
    atomicAdd(&csum[c], s);
}

// ---------------- K4c: score_query = exp(score - colmax) / colsum -------------
__global__ __launch_bounds__(256)
void k_scorequery(const float* __restrict__ score, const unsigned* __restrict__ cmaxu,
                  const float* __restrict__ csum, float* __restrict__ outSQ) {
    size_t i = (size_t)blockIdx.x * 256 + threadIdx.x;
    int c = (int)(i & 511);
    outSQ[i] = expf(score[i] - funflip(cmaxu[c])) / csum[c];
}

// ---------------- K5: weighted scatter-sum into memory slots ------------------
// weight_n = score_query[n,g]/colmax(score_query[:,g]) == exp(score[n,g]-colmax[g])
__global__ __launch_bounds__(256)
void k_scatter(const float* __restrict__ qf, const int* __restrict__ gidx,
               const float* __restrict__ rmax, const unsigned* __restrict__ cmaxu,
               float* __restrict__ qupd) {
    int n = blockIdx.x;
    int g = gidx[n];
    float w = expf(rmax[n] - funflip(cmaxu[g]));
    for (int c = threadIdx.x; c < DIMS; c += 256)
        atomicAdd(&qupd[(size_t)g * DIMS + c], w * qf[(size_t)n * DIMS + c]);
}

// ---------------- K6: updated_memory = l2norm(query_update + keys) ------------
__global__ __launch_bounds__(256)
void k_updmem(const float* __restrict__ qupd, const float* __restrict__ keys,
              float* __restrict__ outUM) {
    __shared__ float red[256];
    __shared__ float sInv;
    int g = blockIdx.x, t = threadIdx.x;
    float a = qupd[(size_t)g * DIMS + t]       + keys[(size_t)g * DIMS + t];
    float b = qupd[(size_t)g * DIMS + t + 256] + keys[(size_t)g * DIMS + t + 256];
    red[t] = a * a + b * b;
    __syncthreads();
    for (int s = 128; s > 0; s >>= 1) {
        if (t < s) red[t] += red[t + s];
        __syncthreads();
    }
    if (t == 0) sInv = 1.f / fmaxf(sqrtf(red[0]), 1e-12f);
    __syncthreads();
    outUM[(size_t)g * DIMS + t]       = a * sInv;
    outUM[(size_t)g * DIMS + t + 256] = b * sInv;
}

// ==============================================================================
extern "C" void kernel_launch(void* const* d_in, const int* in_sizes, int n_in,
                              void* d_out, int out_size, void* d_ws, size_t ws_size,
                              hipStream_t stream) {
    const float* query = (const float*)d_in[0];
    const float* keys  = (const float*)d_in[1];
    float* outF = (float*)d_out;

    char* ws = (char*)d_ws;
    float*          qf    = (float*)(ws + WS_QF);
    unsigned short* qbf   = (unsigned short*)(ws + WS_QBF);
    float*          score = (float*)(ws + WS_SCORE);
    unsigned short* smbf  = (unsigned short*)(ws + WS_SMBF);
    unsigned short* kbf   = (unsigned short*)(ws + WS_KBF);
    unsigned short* ktbf  = (unsigned short*)(ws + WS_KTBF);
    unsigned*       cmaxu = (unsigned*)(ws + WS_CMAX);
    float*          csum  = (float*)(ws + WS_CSUM);
    int*            gidx  = (int*)(ws + WS_GIDX);
    float*          rmax  = (float*)(ws + WS_RMAX);
    float*          qupd  = (float*)(ws + WS_QUPD);

    // zero accumulators (graph-capture safe stream memsets)
    hipMemsetAsync(cmaxu, 0, 512 * sizeof(unsigned), stream);
    hipMemsetAsync(csum,  0, 512 * sizeof(float), stream);
    hipMemsetAsync(qupd,  0, (size_t)MSLOT * DIMS * sizeof(float), stream);
    hipMemsetAsync(outF + OFF_GL, 0, 2 * sizeof(float), stream);

    k_prep_keys<<<1024, 256, 0, stream>>>(keys, kbf, ktbf);
    k_normalize<<<256, 256, 0, stream>>>(query, qf, qbf, outF + OFF_UQ);

    dim3 gg(NSAMP / 128, MSLOT / 64);
    // GEMM1: score = q_flat @ keys^T, fused column-max for axis-0 softmax
    k_gemm_bf16<<<gg, 256, 0, stream>>>(qbf, kbf, score, nullptr, cmaxu, 0);

    k_rowpass<<<NSAMP / 8, 256, 0, stream>>>(score, qf, keys,
                                             outF + OFF_SM, smbf, gidx, rmax,
                                             outF + OFF_GL, outF + OFF_SL);

    k_colsum<<<NSAMP / 256, 512, 0, stream>>>(score, cmaxu, csum);
    k_scorequery<<<(NSAMP * (MSLOT / 256)), 256, 0, stream>>>(score, cmaxu, csum,
                                                              outF + OFF_SQ);

    k_scatter<<<NSAMP, 256, 0, stream>>>(qf, gidx, rmax, cmaxu, qupd);
    k_updmem<<<MSLOT, 256, 0, stream>>>(qupd, keys, outF + OFF_UM);

    // GEMM2: concat_memory = score_memory @ keys -> updated_query[:, 512:]
    k_gemm_bf16<<<gg, 256, 0, stream>>>(smbf, ktbf, nullptr, outF + OFF_UQ,
                                        nullptr, 1);
}
